// Net_81853486727727
// MI455X (gfx1250) — matrix operand order repaired and checked by reading, not compiled, not run on hardware
//
#include <hip/hip_runtime.h>
#include <math.h>

#define B_    100
#define NPER  1000
#define EPER  12000
#define NTOT  (B_ * NPER)    // 100000
#define ETOT  (B_ * EPER)    // 1200000
#define KEEP  800
#define FDIM  64

typedef __attribute__((ext_vector_type(16))) _Float16 v16h;
typedef __attribute__((ext_vector_type(8)))  _Float16 v8h;
typedef __attribute__((ext_vector_type(8)))  float    v8f;

// ---------------------------------------------------------------- degree prep
__global__ void k_fill_deg(float* deg) {
    int i = blockIdx.x * 256 + threadIdx.x;
    if (i < NTOT) deg[i] = 1.0f;   // implicit self-loop weight
}

__global__ void k_deg_accum(const int* __restrict__ ei, const float* __restrict__ ew,
                            float* deg) {
    int e = blockIdx.x * 256 + threadIdx.x;
    if (e < ETOT) atomicAdd(&deg[ei[ETOT + e]], ew[e]);   // col = ei[E + e]
}

__global__ void k_rsqrt(float* deg) {
    int i = blockIdx.x * 256 + threadIdx.x;
    if (i < NTOT) deg[i] = rsqrtf(deg[i]);   // deg -> dinv in place
}

// ---------------------------------------------------------------- conv1 linear (K=3, scalar)
__global__ void k_conv1_lin(const float* __restrict__ x, const float* __restrict__ w,
                            float* __restrict__ hw) {
    int t = blockIdx.x * 256 + threadIdx.x;
    if (t >= NTOT * FDIM) return;
    int i = t >> 6, f = t & 63;
    const float* xr = x + i * 3;
    hw[t] = xr[0] * w[f] + xr[1] * w[64 + f] + xr[2] * w[128 + f];
}

// ---------------------------------------------------------------- self-loop term + bias init
__global__ void k_init_self(const float* __restrict__ hw, const float* __restrict__ dinv,
                            const float* __restrict__ bias, float* __restrict__ agg) {
    int t = blockIdx.x * 256 + threadIdx.x;
    if (t >= NTOT * FDIM) return;
    int i = t >> 6, f = t & 63;
    float di = dinv[i];
    agg[t] = di * di * hw[t] + bias[f];
}

// ---------------------------------------------------------------- edge scatter (msg passing)
// one thread = (edge, feature-quad): float4 gather + 4 f32 global atomics
__global__ void k_edge_scatter(const int* __restrict__ ei, const float* __restrict__ ew,
                               const float* __restrict__ dinv, const float* __restrict__ hw,
                               float* __restrict__ agg) {
    long long t = (long long)blockIdx.x * 256 + threadIdx.x;
    if (t >= (long long)ETOT * 16) return;
    int e = (int)(t >> 4);
    int q = (int)(t & 15);
    int r = ei[e], c = ei[ETOT + e];
    float coef = dinv[r] * ew[e] * dinv[c];
    float4 v = ((const float4*)(hw + (size_t)r * FDIM))[q];
    float* dst = agg + (size_t)c * FDIM + q * 4;
    atomicAdd(dst + 0, coef * v.x);
    atomicAdd(dst + 1, coef * v.y);
    atomicAdd(dst + 2, coef * v.z);
    atomicAdd(dst + 3, coef * v.w);
}

// ---------------------------------------------------------------- relu (+ optional f16 copy)
__global__ void k_relu(float* __restrict__ agg, _Float16* __restrict__ h16) {
    int t = blockIdx.x * 256 + threadIdx.x;
    if (t >= NTOT * FDIM) return;
    float v = agg[t];
    v = v > 0.0f ? v : 0.0f;
    agg[t] = v;
    if (h16) h16[t] = (_Float16)v;
}

// ---------------------------------------------------------------- global mean pool -> x1
__global__ void k_pool_mean(const float* __restrict__ h, float* __restrict__ x1) {
    int b = blockIdx.x, t = threadIdx.x;
    int f = t & 63, g = t >> 6;
    __shared__ float red[256];
    const float* hb = h + (size_t)b * NPER * FDIM;
    float p = 0.0f;
    for (int n = g; n < NPER; n += 4) p += hb[(size_t)n * FDIM + f];
    red[t] = p;
    __syncthreads();
    if (g == 0)
        x1[b * FDIM + f] = (red[f] + red[64 + f] + red[128 + f] + red[192 + f]) * (1.0f / NPER);
}

// ---------------------------------------------------------------- W2 -> fragment-ready f16
// layout: idx = ((nt*2+kh)*32 + lane)*16 + e ; value = W2[kh*32 + (e>>3)*16 + (lane>>4)*8 + (e&7)][nt*16 + (lane&15)]
__global__ void k_w2_frag(const float* __restrict__ w2, _Float16* __restrict__ w2h) {
    int t = blockIdx.x * 256 + threadIdx.x;
    if (t >= 4096) return;
    int e    = t & 15;
    int lane = (t >> 4) & 31;
    int kh   = (t >> 9) & 1;
    int nt   = t >> 10;
    int ks   = (e >> 3) * 16 + (lane >> 4) * 8 + (e & 7);
    int n    = nt * 16 + (lane & 15);
    w2h[t] = (_Float16)w2[(kh * 32 + ks) * 64 + n];
}

// ---------------------------------------------------------------- WMMA GEMM: hw2 = h1(f16) @ W2(f16)
// one wave per 16-node stripe; K=64 as two 16x16x32 WMMAs; 4 n-tiles for 64 outputs
__global__ void k_gemm_wmma(const _Float16* __restrict__ A, const _Float16* __restrict__ Bf,
                            float* __restrict__ C) {
    int wave = (blockIdx.x * blockDim.x + threadIdx.x) >> 5;
    int lane = threadIdx.x & 31;
    int m0 = wave * 16;
    if (m0 >= NTOT) return;                 // wave-uniform: EXEC all-ones inside
    int hl = lane >> 4;
    int mr = lane & 15;
    const _Float16* Arow = A + (size_t)(m0 + mr) * FDIM;
    // ISA 16-bit A 16x32 layout: lanes0-15 hold K={0..7,16..23}, lanes16-31 K={8..15,24..31}
    v8h p00 = *(const v8h*)(Arow + hl * 8);
    v8h p01 = *(const v8h*)(Arow + 16 + hl * 8);
    v8h p10 = *(const v8h*)(Arow + 32 + hl * 8);
    v8h p11 = *(const v8h*)(Arow + 48 + hl * 8);
    v16h a0, a1;
#pragma unroll
    for (int e = 0; e < 8; ++e) {
        a0[e] = p00[e]; a0[8 + e] = p01[e];
        a1[e] = p10[e]; a1[8 + e] = p11[e];
    }
    int n_lo = lane & 15;
#pragma unroll
    for (int nt = 0; nt < 4; ++nt) {
        v16h b0 = *(const v16h*)(Bf + (size_t)((nt * 2 + 0) * 32 + lane) * 16);
        v16h b1 = *(const v16h*)(Bf + (size_t)((nt * 2 + 1) * 32 + lane) * 16);
        v8f c = {};
        c = __builtin_amdgcn_wmma_f32_16x16x32_f16(false, a0, false, b0, (short)0, c, false, false);
        c = __builtin_amdgcn_wmma_f32_16x16x32_f16(false, a1, false, b1, (short)0, c, false, false);
        int n = nt * 16 + n_lo;
#pragma unroll
        for (int r = 0; r < 8; ++r)
            C[(size_t)(m0 + hl * 8 + r) * FDIM + n] = c[r];
    }
}

// ---------------------------------------------------------------- pool_w inverse norm
__global__ void k_invnorm(const float* __restrict__ w, float* __restrict__ out) {
    __shared__ float s[64];
    int t = threadIdx.x;
    s[t] = w[t] * w[t];
    __syncthreads();
    for (int st = 32; st > 0; st >>= 1) { if (t < st) s[t] += s[t + st]; __syncthreads(); }
    if (t == 0) *out = rsqrtf(s[0]);
}

// ---------------------------------------------------------------- scores
__global__ void k_score(const float* __restrict__ h2, const float* __restrict__ pw,
                        const float* __restrict__ invn, float* __restrict__ score) {
    int i = blockIdx.x * 256 + threadIdx.x;
    if (i >= NTOT) return;
    const float* hr = h2 + (size_t)i * FDIM;
    float d = 0.0f;
#pragma unroll
    for (int f = 0; f < FDIM; ++f) d += hr[f] * pw[f];
    score[i] = tanhf(d * (*invn));
}

// ---------------------------------------------------------------- top-K pooling per graph
__global__ void k_topk_pool(const float* __restrict__ score, const float* __restrict__ h2,
                            float* __restrict__ x2) {
    int b = blockIdx.x, t = threadIdx.x;
    __shared__ unsigned keys[NPER];
    __shared__ float    gsc[NPER];
    __shared__ int      redI[256];
    __shared__ float    redF[256];
    __shared__ int      s_taken;
    const float* sc = score + b * NPER;
    for (int i = t; i < NPER; i += 256) {
        unsigned u = __float_as_uint(sc[i]);
        keys[i] = (u & 0x80000000u) ? ~u : (u | 0x80000000u);  // monotonic key
    }
    if (t == 0) s_taken = 0;
    __syncthreads();

    // binary search: largest T with count(key >= T) >= KEEP (uniform across block)
    unsigned long long lo = 0, hi = 0xFFFFFFFFull;
    while (lo < hi) {
        unsigned long long mid = (lo + hi + 1) >> 1;
        unsigned m32 = (unsigned)mid;
        int cnt = 0;
        for (int i = t; i < NPER; i += 256) cnt += (keys[i] >= m32) ? 1 : 0;
        redI[t] = cnt; __syncthreads();
        for (int s = 128; s > 0; s >>= 1) { if (t < s) redI[t] += redI[t + s]; __syncthreads(); }
        int total = redI[0];
        __syncthreads();
        if (total >= KEEP) lo = mid; else hi = mid - 1;
    }
    unsigned T = (unsigned)lo;

    // count strictly greater -> how many == T we still need
    int cg = 0;
    for (int i = t; i < NPER; i += 256) cg += (keys[i] > T) ? 1 : 0;
    redI[t] = cg; __syncthreads();
    for (int s = 128; s > 0; s >>= 1) { if (t < s) redI[t] += redI[t + s]; __syncthreads(); }
    int need = KEEP - redI[0];
    __syncthreads();

    // gated score vector (0 for unselected)
    for (int i = t; i < NPER; i += 256) {
        unsigned k = keys[i];
        float g = 0.0f;
        if (k > T) g = sc[i];
        else if (k == T) {
            int slot = atomicAdd(&s_taken, 1);
            if (slot < need) g = sc[i];
        }
        gsc[i] = g;
    }
    __syncthreads();

    // x2[b][f] = sum_n gsc[n] * h2[n][f] / KEEP  (feature-parallel, 4 node groups)
    int f = t & 63, gq = t >> 6;
    const float* hb = h2 + (size_t)b * NPER * FDIM;
    float p = 0.0f;
    for (int n = gq; n < NPER; n += 4) p += gsc[n] * hb[(size_t)n * FDIM + f];
    redF[t] = p; __syncthreads();
    if (gq == 0)
        x2[b * FDIM + f] = (redF[f] + redF[64 + f] + redF[128 + f] + redF[192 + f]) * (1.0f / KEEP);
}

// ---------------------------------------------------------------- MLP head (100 rows)
__global__ void k_mlp(const float* __restrict__ x1, const float* __restrict__ x2,
                      const float* __restrict__ w1, const float* __restrict__ b1,
                      const float* __restrict__ w2, const float* __restrict__ b2,
                      const float* __restrict__ w3, const float* __restrict__ b3,
                      float* __restrict__ out) {
    int b = blockIdx.x, t = threadIdx.x;   // 64 threads
    __shared__ float z[64], z1[64], z2[32];
    z[t] = x1[b * 64 + t] + x2[b * 64 + t];
    __syncthreads();
    float a = b1[t];
#pragma unroll
    for (int k = 0; k < 64; ++k) a += z[k] * w1[k * 64 + t];
    z1[t] = a > 0.0f ? a : 0.0f;
    __syncthreads();
    if (t < 32) {
        float c = b2[t];
#pragma unroll
        for (int k = 0; k < 64; ++k) c += z1[k] * w2[k * 32 + t];
        z2[t] = c > 0.0f ? c : 0.0f;
    }
    __syncthreads();
    if (t == 0) {
        float s = b3[0];
#pragma unroll
        for (int k = 0; k < 32; ++k) s += z2[k] * w3[k];
        out[b] = 1.0f / (1.0f + expf(-s));
    }
}

// ---------------------------------------------------------------- launch
extern "C" void kernel_launch(void* const* d_in, const int* in_sizes, int n_in,
                              void* d_out, int out_size, void* d_ws, size_t ws_size,
                              hipStream_t stream) {
    const float* x   = (const float*)d_in[0];
    const int*   ei  = (const int*)d_in[1];
    const float* ea  = (const float*)d_in[2];
    // d_in[3] = batch (unused: equal-size graphs)
    const float* c1w = (const float*)d_in[4];
    const float* c1b = (const float*)d_in[5];
    const float* c2w = (const float*)d_in[6];
    const float* c2b = (const float*)d_in[7];
    const float* pw  = (const float*)d_in[8];
    const float* l1w = (const float*)d_in[9];
    const float* l1b = (const float*)d_in[10];
    const float* l2w = (const float*)d_in[11];
    const float* l2b = (const float*)d_in[12];
    const float* l3w = (const float*)d_in[13];
    const float* l3b = (const float*)d_in[14];
    float* out = (float*)d_out;

    char* p = (char*)d_ws;
    auto carve = [&](size_t bytes) -> void* {
        void* r = (void*)p;
        p += (bytes + 255) & ~(size_t)255;
        return r;
    };
    float*    deg  = (float*)carve((size_t)NTOT * 4);            // deg -> dinv
    float*    hw   = (float*)carve((size_t)NTOT * FDIM * 4);     // X@W (reused conv1/conv2)
    float*    agg1 = (float*)carve((size_t)NTOT * FDIM * 4);     // conv1 out -> h1
    float*    agg2 = (float*)carve((size_t)NTOT * FDIM * 4);     // conv2 out -> h2
    _Float16* h1h  = (_Float16*)carve((size_t)NTOT * FDIM * 2);  // h1 in f16 for WMMA
    _Float16* w2h  = (_Float16*)carve(4096 * 2);                 // fragment-ready W2
    float*    x1   = (float*)carve((size_t)B_ * FDIM * 4);
    float*    x2   = (float*)carve((size_t)B_ * FDIM * 4);
    float*    scr  = (float*)carve((size_t)NTOT * 4);
    float*    invn = (float*)carve(4);

    const int tb = 256;
    const int gN  = (NTOT + tb - 1) / tb;
    const int gNF = (NTOT * FDIM + tb - 1) / tb;
    const int gE  = (ETOT + tb - 1) / tb;
    const int gES = (int)(((long long)ETOT * 16 + tb - 1) / tb);

    // degrees / normalization
    k_fill_deg <<<gN,  tb, 0, stream>>>(deg);
    k_deg_accum<<<gE,  tb, 0, stream>>>(ei, ea, deg);
    k_rsqrt    <<<gN,  tb, 0, stream>>>(deg);

    // conv1
    k_conv1_lin   <<<gNF, tb, 0, stream>>>(x, c1w, hw);
    k_init_self   <<<gNF, tb, 0, stream>>>(hw, deg, c1b, agg1);
    k_edge_scatter<<<gES, tb, 0, stream>>>(ei, ea, deg, hw, agg1);
    k_relu        <<<gNF, tb, 0, stream>>>(agg1, h1h);
    k_pool_mean   <<<B_, 256, 0, stream>>>(agg1, x1);

    // conv2 (WMMA GEMM for h1 @ W2)
    k_w2_frag     <<<16, 256, 0, stream>>>(c2w, w2h);
    k_gemm_wmma   <<<(NTOT / 16 + 3) / 4, 128, 0, stream>>>(h1h, w2h, hw);
    k_init_self   <<<gNF, tb, 0, stream>>>(hw, deg, c2b, agg2);
    k_edge_scatter<<<gES, tb, 0, stream>>>(ei, ea, deg, hw, agg2);
    k_relu        <<<gNF, tb, 0, stream>>>(agg2, (_Float16*)nullptr);

    // top-k pooling
    k_invnorm  <<<1, 64, 0, stream>>>(pw, invn);
    k_score    <<<gN, tb, 0, stream>>>(agg2, pw, invn, scr);
    k_topk_pool<<<B_, 256, 0, stream>>>(scr, agg2, x2);

    // MLP head
    k_mlp<<<B_, 64, 0, stream>>>(x1, x2, l1w, l1b, l2w, l2b, l3w, l3b, out);
}
